// FSRCNN_85822036508685
// MI455X (gfx1250) — matrix operand order repaired and checked by reading, not compiled, not run on hardware
//
#include <hip/hip_runtime.h>
#include <hip/hip_bf16.h>

typedef _Float16 f16;
typedef __attribute__((ext_vector_type(16))) _Float16 v16h;
typedef __attribute__((ext_vector_type(8)))  _Float16 v8h;
typedef __attribute__((ext_vector_type(8)))  float    v8f;

#define NB   4
#define Hh   384
#define Ww   384
#define HW   (Hh*Ww)
#define PIX  (NB*HW)      // 589824
#define Dm   32
#define Sm   8
#define HC   96
#define WC   96
#define KTAP 9
#define OFFC 18           // 2*KTAP offset channels per DCN block
#define HOUT 1536

// ---------------- conv1: 3x3, 1 -> 32, + bias + PReLU, store f16 [p][32] ----
__global__ void k_conv1(const float* __restrict__ xin, const float* __restrict__ w,
                        const float* __restrict__ b, const float* __restrict__ pr,
                        f16* __restrict__ h1) {
  int t = blockIdx.x * blockDim.x + threadIdx.x;
  if (t >= PIX * Dm) return;
  int c = t & (Dm - 1);
  int p = t >> 5;
  int n = p / HW; int rem = p - n * HW; int y = rem / Ww; int x0 = rem - y * Ww;
  float acc = b[c];
  #pragma unroll
  for (int ky = 0; ky < 3; ++ky) {
    int yy = y + ky - 1;
    if (yy < 0 || yy >= Hh) continue;
    #pragma unroll
    for (int kx = 0; kx < 3; ++kx) {
      int xx = x0 + kx - 1;
      if (xx < 0 || xx >= Ww) continue;
      acc += xin[n * HW + yy * Ww + xx] * w[c * 9 + ky * 3 + kx];
    }
  }
  float s = pr[c];
  acc = acc >= 0.f ? acc : s * acc;
  h1[p * Dm + c] = (f16)acc;
}

// ---------------- squeeze: 1x1 GEMM [Px32]x[32x8] via WMMA f16 ---------------
__global__ void k_squeeze_wmma(const f16* __restrict__ h1, const float* __restrict__ w,
                               const float* __restrict__ b, const float* __restrict__ pr,
                               float* __restrict__ hs, int tiles_per_wave) {
  int lane = threadIdx.x & 31;
  int wave = (blockIdx.x * blockDim.x + threadIdx.x) >> 5;
  int nCol = lane & 15;
  int g    = lane >> 4;
  // B layout: lane holds column nCol; elems 0..7 -> k=g*8.., elems 8..15 -> k=16+g*8..
  v16h bm;
  #pragma unroll
  for (int e = 0; e < 16; ++e) {
    int k = (e < 8) ? (g * 8 + e) : (16 + g * 8 + (e - 8));
    bm[e] = (nCol < Sm) ? (f16)w[nCol * Dm + k] : (f16)0.f;
  }
  float bias  = (nCol < Sm) ? b[nCol]  : 0.f;
  float slope = (nCol < Sm) ? pr[nCol] : 0.f;
  int m = lane & 15;
  int ntiles = PIX / 16;
  for (int tt = 0; tt < tiles_per_wave; ++tt) {
    int tile = wave * tiles_per_wave + tt;
    if (tile >= ntiles) return;
    int pb = tile * 16;
    // A layout: lane row m; elems 0..7 -> k=g*8.., elems 8..15 -> k=16+g*8..
    const f16* row = h1 + (size_t)(pb + m) * Dm;
    v8h lo = *(const v8h*)(row + g * 8);
    v8h hi = *(const v8h*)(row + 16 + g * 8);
    v16h a;
    #pragma unroll
    for (int e = 0; e < 8; ++e) { a[e] = lo[e]; a[e + 8] = hi[e]; }
    v8f cacc;
    #pragma unroll
    for (int r = 0; r < 8; ++r) cacc[r] = bias;
    cacc = __builtin_amdgcn_wmma_f32_16x16x32_f16(false, a, false, bm,
                                                  (short)0, cacc, false, false);
    if (nCol < Sm) {
      #pragma unroll
      for (int r = 0; r < 8; ++r) {
        int mr = g * 8 + r;
        float v = cacc[r];
        v = v >= 0.f ? v : slope * v;
        hs[(size_t)(pb + mr) * Sm + nCol] = v;
      }
    }
  }
}

// ---------------- 4x4 average pool: [p][8] f32 -> coarse [n][yc][xc][8] ------
__global__ void k_pool(const float* __restrict__ hs, float* __restrict__ pool) {
  int t = blockIdx.x * blockDim.x + threadIdx.x;
  if (t >= NB * HC * WC * Sm) return;
  int c = t & (Sm - 1);
  int q = t >> 3;
  int xc = q % WC; q /= WC;
  int yc = q % HC; int n = q / HC;
  float sum = 0.f;
  #pragma unroll
  for (int dy = 0; dy < 4; ++dy)
    #pragma unroll
    for (int dx = 0; dx < 4; ++dx)
      sum += hs[(size_t)(n * HW + (yc * 4 + dy) * Ww + (xc * 4 + dx)) * Sm + c];
  pool[t] = sum * (1.f / 16.f);
}

// ---------------- offset conv: 3x3 pad1, 8 -> 18, clip [-1,1] ----------------
__global__ void k_offconv(const float* __restrict__ pool, const float* __restrict__ w,
                          const float* __restrict__ b, float* __restrict__ off) {
  int t = blockIdx.x * blockDim.x + threadIdx.x;
  if (t >= NB * HC * WC * OFFC) return;
  int oc = t % OFFC;
  int q  = t / OFFC;
  int xc = q % WC; q /= WC;
  int yc = q % HC; int n = q / HC;
  float acc = b[oc];
  for (int ky = 0; ky < 3; ++ky) {
    int yy = yc + ky - 1;
    if (yy < 0 || yy >= HC) continue;
    for (int kx = 0; kx < 3; ++kx) {
      int xx = xc + kx - 1;
      if (xx < 0 || xx >= WC) continue;
      const float* pv = pool + (size_t)(n * HC * WC + yy * WC + xx) * Sm;
      const float* pw = w + (size_t)(oc * Sm) * 9 + ky * 3 + kx;
      #pragma unroll
      for (int c = 0; c < Sm; ++c) acc += pv[c] * pw[c * 9];
    }
  }
  acc = fminf(1.f, fmaxf(-1.f, acc));
  off[(size_t)(n * HC * WC + yc * WC + xc) * OFFC + oc] = acc;
}

// ---------------- DCN bilinear gather of one tap into half of an A vector ----
// Writes 8 f16 channels (tap t of pixel (n,y,x)) into a[BASE..BASE+7].
template <int BASE>
__device__ __forceinline__ void dcn_gather8(const float* __restrict__ hs,
                                            const float* __restrict__ ob,
                                            int n, int y, int x, int t,
                                            v16h& a) {
  if (t >= KTAP) {
    #pragma unroll
    for (int c = 0; c < Sm; ++c) a[BASE + c] = (f16)0.f;
    return;
  }
  float oy = ob[2 * t + 0];
  float ox = ob[2 * t + 1];
  float fy = (float)(y + t / 3 - 1) + oy;
  float fx = (float)(x + t % 3 - 1) + ox;
  float y0f = floorf(fy), x0f = floorf(fx);
  float ly = fy - y0f, lx = fx - x0f;
  float hy = 1.f - ly, hx = 1.f - lx;
  int y0 = (int)y0f, x0i = (int)x0f;
  float samp[Sm];
  #pragma unroll
  for (int c = 0; c < Sm; ++c) samp[c] = 0.f;
  #pragma unroll
  for (int dy = 0; dy < 2; ++dy) {
    int yi = y0 + dy;
    if (yi < 0 || yi >= Hh) continue;
    float wy = dy ? ly : hy;
    #pragma unroll
    for (int dx = 0; dx < 2; ++dx) {
      int xi = x0i + dx;
      if (xi < 0 || xi >= Ww) continue;
      float wgt = wy * (dx ? lx : hx);
      const float* v = hs + (size_t)(n * HW + yi * Ww + xi) * Sm;
      #pragma unroll
      for (int c = 0; c < Sm; ++c) samp[c] += wgt * v[c];
    }
  }
  #pragma unroll
  for (int c = 0; c < Sm; ++c) a[BASE + c] = (f16)samp[c];
}

// ---------------- deformable 3x3 conv, 8 -> 8, via WMMA ----------------------
// Per 16-pixel tile: K = 9 taps * 8 ch = 72, padded to 96 -> 3 chained WMMAs.
// K ordered tap-major: for chunk j, lane-group g supplies taps 4j+2g (elems 0-7)
// and 4j+2g+1 (elems 8-15); the pixel's 9 real taps split across its 2 lanes.
__global__ void k_dcn_wmma(const float* __restrict__ hs, const float* __restrict__ off,
                           const float* __restrict__ w, const float* __restrict__ b,
                           const float* __restrict__ pr, float* __restrict__ out,
                           int tiles_per_wave) {
  int lane = threadIdx.x & 31;
  int wave = (blockIdx.x * blockDim.x + threadIdx.x) >> 5;
  int nCol = lane & 15;
  int g    = lane >> 4;
  // B chunks: B[k][n] with k = 32j + g*16 + e; tap = k>>3, c = k&7
  v16h bw[3];
  #pragma unroll
  for (int j = 0; j < 3; ++j) {
    #pragma unroll
    for (int e = 0; e < 16; ++e) {
      int k = 32 * j + g * 16 + e;
      int tap = k >> 3, c = k & 7;
      bw[j][e] = (tap < KTAP && nCol < Sm)
                     ? (f16)w[(size_t)(nCol * Sm + c) * 9 + tap]
                     : (f16)0.f;
    }
  }
  float bias  = (nCol < Sm) ? b[nCol]  : 0.f;
  float slope = (nCol < Sm) ? pr[nCol] : 0.f;
  int m = lane & 15;
  int ntiles = PIX / 16;
  for (int tt = 0; tt < tiles_per_wave; ++tt) {
    int tile = wave * tiles_per_wave + tt;
    if (tile >= ntiles) return;
    int pb = tile * 16;                  // 16 consecutive x in one row
    int p = pb + m;
    int n = p / HW; int rem = p - n * HW; int y = rem / Ww; int x = rem - y * Ww;
    const float* ob = off + (size_t)(n * HC * WC + (y >> 2) * WC + (x >> 2)) * OFFC;
    v8f acc;
    #pragma unroll
    for (int r = 0; r < 8; ++r) acc[r] = bias;
    #pragma unroll
    for (int j = 0; j < 3; ++j) {
      v16h a;
      dcn_gather8<0>(hs, ob, n, y, x, 4 * j + 2 * g, a);
      dcn_gather8<8>(hs, ob, n, y, x, 4 * j + 2 * g + 1, a);
      acc = __builtin_amdgcn_wmma_f32_16x16x32_f16(false, a, false, bw[j],
                                                   (short)0, acc, false, false);
    }
    if (nCol < Sm) {
      #pragma unroll
      for (int r = 0; r < 8; ++r) {
        int mr = g * 8 + r;
        float v = acc[r];
        v = v >= 0.f ? v : slope * v;
        out[(size_t)(pb + mr) * Sm + nCol] = v;
      }
    }
  }
}

// ---------------- expand: 1x1, 8 -> 32, + PReLU, store f16 [p][32] -----------
__global__ void k_expand(const float* __restrict__ hs, const float* __restrict__ w,
                         const float* __restrict__ b, const float* __restrict__ pr,
                         f16* __restrict__ he) {
  int t = blockIdx.x * blockDim.x + threadIdx.x;
  if (t >= PIX * Dm) return;
  int oc = t & 31; int p = t >> 5;
  const float* row = hs + (size_t)p * Sm;
  float acc = b[oc];
  #pragma unroll
  for (int c = 0; c < Sm; ++c) acc += row[c] * w[oc * Sm + c];
  float s = pr[oc];
  acc = acc >= 0.f ? acc : s * acc;
  he[t] = (f16)acc;
}

// ---------------- up conv: 3x3, 32 -> 16, WMMA, fused pixel-shuffle ----------
__global__ void k_upconv_wmma(const f16* __restrict__ he, const float* __restrict__ w,
                              const float* __restrict__ b, float* __restrict__ out,
                              int tiles_per_wave) {
  int lane = threadIdx.x & 31;
  int wave = (blockIdx.x * blockDim.x + threadIdx.x) >> 5;
  int nCol = lane & 15;
  int g    = lane >> 4;
  v16h bt[KTAP];
  #pragma unroll
  for (int tap = 0; tap < KTAP; ++tap) {
    #pragma unroll
    for (int e = 0; e < 16; ++e) {
      int k = (e < 8) ? (g * 8 + e) : (16 + g * 8 + (e - 8));  // B: table order
      bt[tap][e] = (f16)w[(size_t)(nCol * Dm + k) * 9 + tap];
    }
  }
  float bias = b[nCol];
  int ry = nCol >> 2, rx = nCol & 3;           // pixel-shuffle sub-position
  int m = lane & 15;
  int ntiles = PIX / 16;
  for (int tt = 0; tt < tiles_per_wave; ++tt) {
    int tile = wave * tiles_per_wave + tt;
    if (tile >= ntiles) return;
    int pb = tile * 16;                        // 16 consecutive x in one row
    int n = pb / HW; int rem = pb - n * HW; int y = rem / Ww; int x0 = rem - y * Ww;
    // prefetch next tile's center row (lights up global_prefetch on gfx1250)
    if (tile + 1 < ntiles)
      __builtin_prefetch(he + (size_t)(pb + 16) * Dm, 0, 1);
    v8f acc;
    #pragma unroll
    for (int r = 0; r < 8; ++r) acc[r] = bias;
    #pragma unroll
    for (int tap = 0; tap < KTAP; ++tap) {
      int ky = tap / 3, kx = tap % 3;
      int yy = y + ky - 1;
      int xx = x0 + m + kx - 1;
      v8h lo = {};
      v8h hi = {};
      if (yy >= 0 && yy < Hh && xx >= 0 && xx < Ww) {
        const f16* row = he + (size_t)(n * HW + yy * Ww + xx) * Dm;
        lo = *(const v8h*)(row + g * 8);
        hi = *(const v8h*)(row + 16 + g * 8);
      }
      v16h a;
      #pragma unroll
      for (int e = 0; e < 8; ++e) { a[e] = lo[e]; a[e + 8] = hi[e]; }
      acc = __builtin_amdgcn_wmma_f32_16x16x32_f16(false, a, false, bt[tap],
                                                   (short)0, acc, false, false);
    }
    #pragma unroll
    for (int r = 0; r < 8; ++r) {
      int mr = g * 8 + r;
      int px = x0 + mr;
      out[(size_t)n * (HOUT * HOUT) + (size_t)(y * 4 + ry) * HOUT + (px * 4 + rx)] = acc[r];
    }
  }
}

// ---------------- host orchestration -----------------------------------------
extern "C" void kernel_launch(void* const* d_in, const int* in_sizes, int n_in,
                              void* d_out, int out_size, void* d_ws, size_t ws_size,
                              hipStream_t stream) {
  (void)in_sizes; (void)n_in; (void)out_size; (void)ws_size;
  const float* x       = (const float*)d_in[0];
  const float* conv1_w = (const float*)d_in[1];
  const float* conv1_b = (const float*)d_in[2];
  const float* prelu1  = (const float*)d_in[3];
  const float* sq_w    = (const float*)d_in[4];
  const float* sq_b    = (const float*)d_in[5];
  const float* prelu_s = (const float*)d_in[6];
  const float* off_w   = (const float*)d_in[7];
  const float* off_b   = (const float*)d_in[8];
  const float* dcn_w   = (const float*)d_in[9];
  const float* dcn_b   = (const float*)d_in[10];
  const float* prelu_d = (const float*)d_in[11];
  const float* ex_w    = (const float*)d_in[12];
  const float* ex_b    = (const float*)d_in[13];
  const float* prelu_e = (const float*)d_in[14];
  const float* up_w    = (const float*)d_in[15];
  const float* up_b    = (const float*)d_in[16];
  float* out = (float*)d_out;

  // Workspace layout (~117 MB total):
  char*  ws   = (char*)d_ws;
  f16*   h1   = (f16*)ws;                                   // PIX*32 f16
  float* hs0  = (float*)(ws + (size_t)PIX * Dm * sizeof(f16));
  float* hs1  = hs0 + (size_t)PIX * Sm;
  float* pool = hs1 + (size_t)PIX * Sm;
  float* off  = pool + (size_t)NB * HC * WC * Sm;
  f16*   he   = (f16*)(off + (size_t)NB * HC * WC * OFFC);  // 16B-aligned

  const int threads = 256;
  const int ntiles  = PIX / 16;

  k_conv1<<<(PIX * Dm + 255) / 256, threads, 0, stream>>>(x, conv1_w, conv1_b, prelu1, h1);

  {
    int tpw = 8;
    int waves = (ntiles + tpw - 1) / tpw;
    int blocks = (waves * 32 + 255) / 256;
    k_squeeze_wmma<<<blocks, threads, 0, stream>>>(h1, sq_w, sq_b, prelu_s, hs0, tpw);
  }

  float* cur = hs0; float* nxt = hs1;
  for (int i = 0; i < 4; ++i) {
    k_pool<<<(NB * HC * WC * Sm + 255) / 256, threads, 0, stream>>>(cur, pool);
    k_offconv<<<(NB * HC * WC * OFFC + 255) / 256, threads, 0, stream>>>(
        pool, off_w + (size_t)i * OFFC * Sm * 9, off_b + i * OFFC, off);
    {
      int tpw = 4;                              // heavy tiles: more waves in flight
      int waves = (ntiles + tpw - 1) / tpw;
      int blocks = (waves * 32 + 255) / 256;
      k_dcn_wmma<<<blocks, threads, 0, stream>>>(
          cur, off, dcn_w + (size_t)i * Sm * Sm * 9, dcn_b + i * Sm, prelu_d + i * Sm,
          nxt, tpw);
    }
    float* t2 = cur; cur = nxt; nxt = t2;
  }

  k_expand<<<(PIX * Dm + 255) / 256, threads, 0, stream>>>(cur, ex_w, ex_b, prelu_e, he);

  {
    int tpw = 8;
    int waves = (ntiles + tpw - 1) / tpw;
    int blocks = (waves * 32 + 255) / 256;
    k_upconv_wmma<<<blocks, threads, 0, stream>>>(he, up_w, up_b, out, tpw);
  }
}
